// GaussianDenoiser_87643102642782
// MI455X (gfx1250) — matrix operand order repaired and checked by reading, not compiled, not run on hardware
//
#include <hip/hip_runtime.h>
#include <hip/hip_bf16.h>
#include <stdint.h>

// ---- CDNA5 WMMA types -------------------------------------------------------
typedef __attribute__((ext_vector_type(16))) __bf16        v16bf;
typedef __attribute__((ext_vector_type(8)))  float         v8f;
typedef __attribute__((ext_vector_type(4)))  unsigned int  u32x4;

union Op16 { u32x4 q[2]; v16bf v; };   // 32 bytes = one 16-bit WMMA operand

#define KC 64   // K per staging phase = two bf16 WMMA K-steps

// ---------------------------------------------------------------------------
// Kernel 1: TWO batches per workgroup.
//   cap[b] = I + V^T diag(inv_d[b]) U   (WMMA bf16, f32 accumulate)
//   w[b]   = V^T z[b]                   (fused: z in col 0 of an extra B tile)
// then two concurrent in-LDS Gaussian eliminations produce sol[b0], sol[b1].
// 256 threads = 8 wave32. Raw U/V chunks are loaded ONCE and scaled twice
// (inv_d differs per batch only), halving L2 traffic per batch; the shared
// A operand (V^T tile) feeds 5 WMMAs per wave per K-step with no divergence.
// inv_d for chunk kk+KC is computed during the WMMA phase (double-buffered):
// 2 barriers per 64-wide K chunk.
// ---------------------------------------------------------------------------
__global__ __launch_bounds__(256)
void gd_cap_solve_kernel(const float* __restrict__ xt,
                         const float* __restrict__ sigma_t,
                         const float* __restrict__ mu_x,
                         const float* __restrict__ diag,
                         const float* __restrict__ U,
                         const float* __restrict__ V,
                         float* __restrict__ sol_out,
                         int D)
{
    __shared__ __align__(16) __bf16 Vt [64][80];  // A source: Vt[col][k], shared
    __shared__ __align__(16) __bf16 Ut0[64][80];  // B source: inv_d[b0] * U
    __shared__ __align__(16) __bf16 Ut1[64][80];  // B source: inv_d[b1] * U
    __shared__ __align__(16) __bf16 Zt0[16][80];  // w B-tile, col 0 = z[b0] chunk
    __shared__ __align__(16) __bf16 Zt1[16][80];  // w B-tile, col 0 = z[b1] chunk
    __shared__ float invd2[2][2][KC];             // [buf][batch][k]
    __shared__ float capL[2][64][66];             // two 64x64 | rhs col 64 | pad
    __shared__ float solL[2][64];

    const int tid  = threadIdx.x;
    const int lane = tid & 31;
    const int wv   = tid >> 5;
    const int b0   = blockIdx.x * 2;
    const int b1   = b0 + 1;

    const float covt0 = sigma_t[b0] * sigma_t[b0];
    const float covt1 = sigma_t[b1] * sigma_t[b1];

    // Zero Zt0/Zt1 once: rows 1..15 stay zero (only col 0 carries z).
    for (int i = tid; i < 16 * 80; i += 256) {
        (&Zt0[0][0])[i] = (__bf16)0.0f;
        (&Zt1[0][0])[i] = (__bf16)0.0f;
    }
    // Prologue: inv_d for the first chunk (loop-top barrier orders it).
    if (tid < KC) {
        const float dg = diag[tid];
        invd2[0][0][tid] = 1.0f / (dg + covt0);
        invd2[0][1][tid] = 1.0f / (dg + covt1);
    }

    // Tile assignment: wave wv -> cap row-tile r0 = wv&3, col-tiles {s0,s0+1}
    // for BOTH batches; w tile for batch h = wv>>2, row-tile r0.
    const int r0 = wv & 3;
    const int h  = wv >> 2;
    const int s0 = h * 2;
    const int s1 = s0 + 1;

    v8f accA0 = {}; v8f accB0 = {};   // batch 0 cap tiles (r0,s0) (r0,s1)
    v8f accA1 = {}; v8f accB1 = {};   // batch 1 cap tiles
    v8f accW  = {};                   // w tile, batch h, row-tile r0

    const int    m    = lane & 15;
    const bool   lo   = lane < 16;    // K-half selector (contiguous k = 16h + s, A and B alike)
    const size_t row0 = (size_t)b0 * (size_t)D;
    const size_t row1 = (size_t)b1 * (size_t)D;

    __bf16 (* const Zth)[80] = h ? Zt1 : Zt0;   // wave-uniform

    for (int kk = 0; kk < D; kk += KC) {
        const int cur = (kk >> 6) & 1;
        const float* ic0 = invd2[cur][0];
        const float* ic1 = invd2[cur][1];

        __syncthreads();   // LDS reuse fence (also orders invd2[cur] producer)

        // Stage 64x64 chunk: raw U/V loaded ONCE, U scaled per batch, all bf16,
        // transposed. Global reads coalesced over the 64-wide row dimension.
        #pragma unroll 8
        for (int i = 0; i < 16; ++i) {
            const int f  = i * 256 + tid;
            const int dl = f >> 6;             // 0..63  (k within chunk)
            const int c  = f & 63;             // 0..63  (column r/s)
            const size_t g = (size_t)(kk + dl) * 64 + c;
            const float uv = U[g];
            Vt [c][dl] = (__bf16)V[g];
            Ut0[c][dl] = (__bf16)(uv * ic0[dl]);
            Ut1[c][dl] = (__bf16)(uv * ic1[dl]);
        }
        if (tid < KC) {
            const int d = kk + tid;
            const float mv = mu_x[d];
            Zt0[0][tid] = (__bf16)((xt[row0 + d] - mv) * ic0[tid]);
            Zt1[0][tid] = (__bf16)((xt[row1 + d] - mv) * ic1[tid]);
        }
        __syncthreads();

        // Pipelined: inv_d for the next chunk + xt prefetch, by waves 0-1
        // (tid<64 is wave-uniform -> EXEC all-ones around every WMMA).
        if (tid < KC && kk + KC < D) {
            const int d = kk + KC + tid;
            const float dg = diag[d];
            invd2[cur ^ 1][0][tid] = 1.0f / (dg + covt0);
            invd2[cur ^ 1][1][tid] = 1.0f / (dg + covt1);
            __builtin_prefetch(&xt[row0 + d], 0, 1);
            __builtin_prefetch(&xt[row1 + d], 0, 1);
        }

        // Two WMMA K-steps; 5 WMMAs per wave per step, shared A operand.
        #pragma unroll
        for (int t = 0; t < 2; ++t) {
            const int kb = t * 32;
            const int co = kb + (lo ? 0 : 16);

            // A operand (16x32 bf16): lane m<16 -> K 0..15,
            // lane m+16 -> K 16..31.
            Op16 a;
            const __bf16* ar = &Vt[r0 * 16 + m][kb];
            a.q[0] = *reinterpret_cast<const u32x4*>(ar + (lo ? 0 : 16));
            a.q[1] = *reinterpret_cast<const u32x4*>(ar + (lo ? 8 : 24));

            // B operands (32x16 bf16): lane = column n, contiguous K half.
            Op16 p00, p01, p10, p11, pz;
            const __bf16* c00 = &Ut0[s0 * 16 + m][co];
            p00.q[0] = *reinterpret_cast<const u32x4*>(c00);
            p00.q[1] = *reinterpret_cast<const u32x4*>(c00 + 8);
            const __bf16* c01 = &Ut0[s1 * 16 + m][co];
            p01.q[0] = *reinterpret_cast<const u32x4*>(c01);
            p01.q[1] = *reinterpret_cast<const u32x4*>(c01 + 8);
            const __bf16* c10 = &Ut1[s0 * 16 + m][co];
            p10.q[0] = *reinterpret_cast<const u32x4*>(c10);
            p10.q[1] = *reinterpret_cast<const u32x4*>(c10 + 8);
            const __bf16* c11 = &Ut1[s1 * 16 + m][co];
            p11.q[0] = *reinterpret_cast<const u32x4*>(c11);
            p11.q[1] = *reinterpret_cast<const u32x4*>(c11 + 8);
            const __bf16* cz = &Zth[m][co];
            pz.q[0] = *reinterpret_cast<const u32x4*>(cz);
            pz.q[1] = *reinterpret_cast<const u32x4*>(cz + 8);

            accA0 = __builtin_amdgcn_wmma_f32_16x16x32_bf16(false, a.v, false, p00.v,
                                                            (short)0, accA0, false, false);
            accB0 = __builtin_amdgcn_wmma_f32_16x16x32_bf16(false, a.v, false, p01.v,
                                                            (short)0, accB0, false, false);
            accA1 = __builtin_amdgcn_wmma_f32_16x16x32_bf16(false, a.v, false, p10.v,
                                                            (short)0, accA1, false, false);
            accB1 = __builtin_amdgcn_wmma_f32_16x16x32_bf16(false, a.v, false, p11.v,
                                                            (short)0, accB1, false, false);
            accW  = __builtin_amdgcn_wmma_f32_16x16x32_bf16(false, a.v, false, pz.v,
                                                            (short)0, accW, false, false);
        }
    }

    __syncthreads();
    // Spill accumulators: C/D layout = lane n (col), VGPR v -> row v (+8, lanes>=16)
    {
        const int n  = lane & 15;
        const int mb = lo ? 0 : 8;
        #pragma unroll
        for (int v = 0; v < 8; ++v) {
            capL[0][r0 * 16 + mb + v][s0 * 16 + n] = accA0[v];
            capL[0][r0 * 16 + mb + v][s1 * 16 + n] = accB0[v];
            capL[1][r0 * 16 + mb + v][s0 * 16 + n] = accA1[v];
            capL[1][r0 * 16 + mb + v][s1 * 16 + n] = accB1[v];
        }
        if (n == 0) {
            #pragma unroll
            for (int v = 0; v < 8; ++v)
                capL[h][r0 * 16 + mb + v][64] = accW[v];   // rhs = w[batch h]
        }
    }
    __syncthreads();
    if (tid < 128) capL[tid >> 6][tid & 63][tid & 63] += 1.0f;   // cap = I + Gram

    // Two concurrent Gaussian eliminations (no pivoting: cap ~ I + O(1/sqrt(D))).
    // Threads 0-63 own system 0's rows; threads 64-127 own system 1's rows.
    for (int k = 0; k < 63; ++k) {
        __syncthreads();
        if (tid < 128) {
            const int sys = tid >> 6;
            const int row = tid & 63;
            if (row > k) {
                const float fct = capL[sys][row][k] / capL[sys][k][k];
                for (int j = k + 1; j <= 64; ++j)
                    capL[sys][row][j] -= fct * capL[sys][k][j];
            }
        }
    }
    // Back substitution (both systems in lockstep).
    for (int k = 63; k >= 0; --k) {
        __syncthreads();
        if (tid < 128 && (tid & 63) == k) {
            const int sys = tid >> 6;
            solL[sys][k] = capL[sys][k][64] / capL[sys][k][k];
        }
        __syncthreads();
        if (tid < 128 && (tid & 63) < k) {
            const int sys = tid >> 6;
            capL[sys][tid & 63][64] -= capL[sys][tid & 63][k] * solL[sys][k];
        }
    }
    __syncthreads();
    if (tid < 128)
        sol_out[(size_t)(b0 + (tid >> 6)) * 64 + (tid & 63)] = solL[tid >> 6][tid & 63];
}

// ---------------------------------------------------------------------------
// Kernel 2: out = xt - covt * ( z - inv_d * (U @ sol) ), single xt stream.
// U rows come from L2 (U is 4 MB, fully L2-resident). HBM-stream-bound.
// ---------------------------------------------------------------------------
__global__ __launch_bounds__(256)
void gd_apply_kernel(const float* __restrict__ xt,
                     const float* __restrict__ sigma_t,
                     const float* __restrict__ mu_x,
                     const float* __restrict__ diag,
                     const float* __restrict__ U,
                     const float* __restrict__ sol,
                     float* __restrict__ out,
                     int D)
{
    __shared__ float s_sol[64];
    const int tid = threadIdx.x;
    const int b   = blockIdx.y;
    const int d0  = blockIdx.x * 2048;

    if (tid < 64) s_sol[tid] = sol[(size_t)b * 64 + tid];
    __syncthreads();

    const float sg   = sigma_t[b];
    const float covt = sg * sg;
    const size_t rowBase = (size_t)b * (size_t)D;

    #pragma unroll 2
    for (int i = 0; i < 8; ++i) {
        const int d = d0 + i * 256 + tid;
        const float xv = xt[rowBase + d];
        const float iv = 1.0f / (diag[d] + covt);
        float us = 0.0f;
        const float4* Ur = reinterpret_cast<const float4*>(U + (size_t)d * 64);
        #pragma unroll
        for (int r4 = 0; r4 < 16; ++r4) {
            const float4 u = Ur[r4];
            us += u.x * s_sol[4 * r4 + 0] + u.y * s_sol[4 * r4 + 1]
                + u.z * s_sol[4 * r4 + 2] + u.w * s_sol[4 * r4 + 3];
        }
        const float z = (xv - mu_x[d]) * iv;
        const float y = z - iv * us;
        out[rowBase + d] = xv - covt * y;
    }
}

// ---------------------------------------------------------------------------
extern "C" void kernel_launch(void* const* d_in, const int* in_sizes, int n_in,
                              void* d_out, int out_size, void* d_ws, size_t ws_size,
                              hipStream_t stream)
{
    (void)n_in; (void)out_size; (void)ws_size;
    const float* xt      = (const float*)d_in[0];
    const float* sigma_t = (const float*)d_in[1];
    const float* mu_x    = (const float*)d_in[2];
    const float* diag    = (const float*)d_in[3];
    const float* U       = (const float*)d_in[4];
    const float* V       = (const float*)d_in[5];
    float* out = (float*)d_out;

    const int B = in_sizes[1];   // sigma_t has B elements (1024, even)
    const int D = in_sizes[2];   // mu_x has D elements (16384)

    float* sol = (float*)d_ws;   // B * 64 floats = 256 KB of workspace

    gd_cap_solve_kernel<<<dim3(B / 2), dim3(256), 0, stream>>>(
        xt, sigma_t, mu_x, diag, U, V, sol, D);

    gd_apply_kernel<<<dim3(D / 2048, B), dim3(256), 0, stream>>>(
        xt, sigma_t, mu_x, diag, U, sol, out, D);
}
